// CausalSelfAttention_14912126451796
// MI455X (gfx1250) — matrix operand order repaired and checked by reading, not compiled, not run on hardware
//
#include <hip/hip_runtime.h>
#include <hip/hip_bf16.h>
#include <stdint.h>

#define BB 4
#define TT 2048
#define CC 1024
#define HH 16
#define DD 64
#define C3 3072

typedef __bf16 bf16;
typedef __attribute__((ext_vector_type(16))) __bf16 v16bf;
typedef __attribute__((ext_vector_type(8)))  __bf16 v8bf;
typedef __attribute__((ext_vector_type(8)))  float  v8f;
typedef __attribute__((ext_vector_type(4)))  uint32_t u32x4;
typedef __attribute__((ext_vector_type(8)))  uint32_t u32x8;

// LDS row stride for TDM-staged B tiles: 64 elems (128B) + 8 elems (16B) TDM pad
#define BROW 72

__device__ __forceinline__ v8f wmma_bf16(v16bf a, v16bf b, v8f c) {
  return __builtin_amdgcn_wmma_f32_16x16x32_bf16(false, a, false, b, (short)0, c, false, false);
}

// A-operand 16x32 bf16 from row-major [M][K] storage.
__device__ __forceinline__ v16bf load_a16x32(const bf16* __restrict__ base, int ld) {
  int lane = threadIdx.x & 31;
  int row  = lane & 15;
  int koff = (lane >> 4) << 3;
  const bf16* p = base + (size_t)row * ld + koff;
  v8bf lo = *(const v8bf*)(p);
  v8bf hi = *(const v8bf*)(p + 16);
  v16bf r;
#pragma unroll
  for (int i = 0; i < 8; ++i) { r[i] = lo[i]; r[i + 8] = hi[i]; }
  return r;
}

// Tensor Data Mover: 2-D tile (tile_h rows x tile_w bf16) from a row-major
// bf16 tensor (width tensor_w, height tensor_h, row stride row_stride elems)
// into LDS at lds_addr, with 4 dwords of LDS pad after every 32 dwords (one row).
__device__ __forceinline__ void tdm_load_b_tile(uint32_t lds_addr, const bf16* gptr,
                                                uint32_t tensor_w, uint32_t tensor_h,
                                                uint32_t tile_w, uint32_t tile_h,
                                                uint32_t row_stride) {
  uint64_t ga = (uint64_t)(uintptr_t)gptr;
  u32x4 g0;
  g0[0] = 1u;                                               // count=1, user mode
  g0[1] = lds_addr;                                         // LDS byte address
  g0[2] = (uint32_t)ga;                                     // global_addr[31:0]
  g0[3] = (uint32_t)((ga >> 32) & 0x1FFFFFFu) | (2u << 30); // global_addr[56:32] | type=2
  u32x8 g1;
  g1[0] = (1u << 16)              // data_size = 1 (2 bytes)
        | (1u << 20)              // pad_enable
        | (4u << 22)              // pad_interval: 32 dwords (one 64-elem row)
        | (3u << 25);             // pad_amount: 4 dwords (16B) per row
  g1[1] = (tensor_w & 0xFFFFu) << 16;                          // tensor_dim0 lo16
  g1[2] = (tensor_w >> 16) | ((tensor_h & 0xFFFFu) << 16);     // dim0 hi | dim1 lo
  g1[3] = (tensor_h >> 16) | ((tile_w & 0xFFFFu) << 16);       // dim1 hi | tile_dim0
  g1[4] = (tile_h & 0xFFFFu);                                  // tile_dim1, tile_dim2=0
  g1[5] = row_stride;                                          // tensor_dim0_stride lo32
  g1[6] = 0;                                                   // stride hi / dim1_stride lo
  g1[7] = 0;
  asm volatile("tensor_load_to_lds %0, %1" :: "s"(g0), "s"(g1) : "memory");
}

// ---------------- fp32 -> bf16 convert ----------------
__global__ void f32_to_bf16(const float* __restrict__ src, bf16* __restrict__ dst, int n) {
  int i = blockIdx.x * blockDim.x + threadIdx.x;
  int stride = gridDim.x * blockDim.x;
  for (; i < n; i += stride) dst[i] = (bf16)src[i];
}

// ---------------- QKV GEMM: [B*T, C] x [C, 3C], TDM-staged B, scatter q/k^T/v --------
__global__ void __launch_bounds__(256) qkv_gemm(const bf16* __restrict__ xb,
                                                const bf16* __restrict__ wb,
                                                bf16* __restrict__ qb,
                                                bf16* __restrict__ ktb,
                                                bf16* __restrict__ vb) {
  __shared__ bf16 bbuf[2][32 * BROW];
  const int BN = C3 / 64;                 // 48 N-blocks
  int bm   = blockIdx.x / BN;
  int bn   = blockIdx.x % BN;
  int warp = threadIdx.x >> 5;
  int lane = threadIdx.x & 31;
  int m0 = bm * 128 + warp * 16;
  int n0 = bn * 64;

  const int KSTEPS = CC / 32;
  if (warp == 0)
    tdm_load_b_tile((uint32_t)(uintptr_t)&bbuf[0][0], wb + n0, C3, CC, 64, 32, C3);

  v8f acc[4] = {};
  for (int i = 0; i < KSTEPS; ++i) {
    if (warp == 0) __builtin_amdgcn_s_wait_tensorcnt(0);
    __syncthreads();                      // publish buf[i&1]; readers of buf[(i+1)&1] done last iter
    if (warp == 0 && (i + 1) < KSTEPS)
      tdm_load_b_tile((uint32_t)(uintptr_t)&bbuf[(i + 1) & 1][0],
                      wb + (size_t)(i + 1) * 32 * C3 + n0, C3, CC, 64, 32, C3);

    v16bf a = load_a16x32(xb + (size_t)m0 * CC + i * 32, CC);
    const bf16* brow = &bbuf[i & 1][(lane & 15) + ((lane >> 4) << 3) * BROW];   // B-operand: lane%16 = N col, K gathered
#pragma unroll
    for (int s = 0; s < 4; ++s) {
      v16bf bmv;
      for (int e = 0; e < 8; ++e) { bmv[e] = brow[e * BROW + s * 16]; bmv[e + 8] = brow[(16 + e) * BROW + s * 16]; }
      acc[s] = wmma_bf16(a, bmv, acc[s]);
    }
  }

  int half = lane >> 4;
  int col  = lane & 15;
#pragma unroll
  for (int s = 0; s < 4; ++s) {
#pragma unroll
    for (int r = 0; r < 8; ++r) {
      int m = m0 + r + 8 * half;          // token index over B*T
      int n = n0 + s * 16 + col;          // [0, 3C)
      int sel = n >> 10;                  // 0=q 1=k 2=v
      int c = n & (CC - 1);
      int h = c >> 6, d = c & 63;
      int bi = m >> 11;
      int t  = m & (TT - 1);
      int bh = bi * HH + h;
      float v = acc[s][r];
      if (sel == 0)       qb [((size_t)bh * TT + t) * DD + d] = (bf16)(v * 0.125f);
      else if (sel == 1)  ktb[((size_t)bh * DD + d) * TT + t] = (bf16)v;
      else                vb [((size_t)bh * TT + t) * DD + d] = (bf16)v;
    }
  }
}

// ---------------- Flash attention: one wave per (b,h, 16-row q tile) ----------------
#define LDSP 40
__global__ void __launch_bounds__(128) attn_kernel(const bf16* __restrict__ qb,
                                                   const bf16* __restrict__ ktb,
                                                   const bf16* __restrict__ vb,
                                                   bf16* __restrict__ yb) {
  __shared__ bf16 lds[4][16 * LDSP];
  int warp = threadIdx.x >> 5;
  int lane = threadIdx.x & 31;
  int wid  = blockIdx.x * 4 + warp;
  int qtile = wid & 127;
  int bh    = wid >> 7;
  int qbase = qtile * 16;

  const bf16* qptr = qb + ((size_t)bh * TT + qbase) * DD;
  v16bf aq0 = load_a16x32(qptr, DD);
  v16bf aq1 = load_a16x32(qptr + 32, DD);

  float mrow[8], lrow[8];
  v8f o[4] = {};
#pragma unroll
  for (int r = 0; r < 8; ++r) { mrow[r] = -1e30f; lrow[r] = 0.f; }

  int half = lane >> 4;
  int col  = lane & 15;
  volatile bf16* lp = lds[warp];

  for (int kb = 0; kb < qbase + 16; kb += 32) {
    const bf16* kt0 = ktb + (size_t)bh * DD * TT + kb;
    v8f s0 = {}, s1 = {};
    {
      const bf16* kp = kt0 + col + (size_t)(half << 3) * TT;
      v16bf b00, b01, b10, b11;
      for (int e = 0; e < 8; ++e) {
        b00[e] = kp[(size_t)e * TT];             b00[e + 8] = kp[(size_t)(16 + e) * TT];
        b01[e] = kp[(size_t)(32 + e) * TT];      b01[e + 8] = kp[(size_t)(48 + e) * TT];
        b10[e] = kp[16 + (size_t)e * TT];        b10[e + 8] = kp[16 + (size_t)(16 + e) * TT];
        b11[e] = kp[16 + (size_t)(32 + e) * TT]; b11[e + 8] = kp[16 + (size_t)(48 + e) * TT];
      }
      s0 = wmma_bf16(aq0, b00, s0);
      s0 = wmma_bf16(aq1, b01, s0);
      s1 = wmma_bf16(aq0, b10, s1);
      s1 = wmma_bf16(aq1, b11, s1);
    }
    float pm[8];
#pragma unroll
    for (int r = 0; r < 8; ++r) {
      int rowg = qbase + r + 8 * half;
      if (kb + col > rowg)      s0[r] = -1e30f;
      if (kb + 16 + col > rowg) s1[r] = -1e30f;
      pm[r] = fmaxf(s0[r], s1[r]);
    }
#pragma unroll
    for (int off = 1; off < 16; off <<= 1)
#pragma unroll
      for (int r = 0; r < 8; ++r) pm[r] = fmaxf(pm[r], __shfl_xor(pm[r], off, 32));

    float fac[8];
#pragma unroll
    for (int r = 0; r < 8; ++r) {
      float mn = fmaxf(mrow[r], pm[r]);
      fac[r] = __expf(mrow[r] - mn);
      mrow[r] = mn;
      s0[r] = __expf(s0[r] - mn);
      s1[r] = __expf(s1[r] - mn);
    }
    float rs[8];
#pragma unroll
    for (int r = 0; r < 8; ++r) rs[r] = s0[r] + s1[r];
#pragma unroll
    for (int off = 1; off < 16; off <<= 1)
#pragma unroll
      for (int r = 0; r < 8; ++r) rs[r] += __shfl_xor(rs[r], off, 32);

#pragma unroll
    for (int r = 0; r < 8; ++r) {
      lrow[r] = lrow[r] * fac[r] + rs[r];
#pragma unroll
      for (int dt = 0; dt < 4; ++dt) o[dt][r] *= fac[r];
      int row = r + 8 * half;
      lp[row * LDSP + col]      = (bf16)s0[r];
      lp[row * LDSP + 16 + col] = (bf16)s1[r];
    }
    v16bf ap;
    {
      int row  = lane & 15;
      int koff = half << 3;
#pragma unroll
      for (int e = 0; e < 8; ++e) {
        ap[e]     = lp[row * LDSP + koff + e];
        ap[e + 8] = lp[row * LDSP + 16 + koff + e];
      }
    }
    const bf16* vp = vb + ((size_t)bh * TT + kb + (half << 3)) * DD + col;
#pragma unroll
    for (int dt = 0; dt < 4; ++dt) {
      v16bf bv;
      for (int e = 0; e < 8; ++e) { bv[e] = vp[e * DD + dt * 16]; bv[e + 8] = vp[(16 + e) * DD + dt * 16]; }
      o[dt] = wmma_bf16(ap, bv, o[dt]);
    }
  }

  int b = bh >> 4, h = bh & 15;
#pragma unroll
  for (int r = 0; r < 8; ++r) {
    float inv = lrow[r] > 0.f ? 1.f / lrow[r] : 0.f;
    int t = qbase + r + 8 * half;
    bf16* yp = yb + ((size_t)b * TT + t) * CC + h * DD;
#pragma unroll
    for (int dt = 0; dt < 4; ++dt) yp[dt * 16 + col] = (bf16)(o[dt][r] * inv);
  }
}

// ---------------- Projection GEMM: [B*T, C] x [C, C], TDM-staged B, fp32 out --------
__global__ void __launch_bounds__(256) proj_gemm(const bf16* __restrict__ yb,
                                                 const bf16* __restrict__ wb,
                                                 float* __restrict__ out) {
  __shared__ bf16 bbuf[2][32 * BROW];
  const int BN = CC / 64;                 // 16 N-blocks
  int bm   = blockIdx.x / BN;
  int bn   = blockIdx.x % BN;
  int warp = threadIdx.x >> 5;
  int lane = threadIdx.x & 31;
  int m0 = bm * 128 + warp * 16;
  int n0 = bn * 64;

  const int KSTEPS = CC / 32;
  if (warp == 0)
    tdm_load_b_tile((uint32_t)(uintptr_t)&bbuf[0][0], wb + n0, CC, CC, 64, 32, CC);

  v8f acc[4] = {};
  for (int i = 0; i < KSTEPS; ++i) {
    if (warp == 0) __builtin_amdgcn_s_wait_tensorcnt(0);
    __syncthreads();
    if (warp == 0 && (i + 1) < KSTEPS)
      tdm_load_b_tile((uint32_t)(uintptr_t)&bbuf[(i + 1) & 1][0],
                      wb + (size_t)(i + 1) * 32 * CC + n0, CC, CC, 64, 32, CC);

    v16bf a = load_a16x32(yb + (size_t)m0 * CC + i * 32, CC);
    const bf16* brow = &bbuf[i & 1][(lane & 15) + ((lane >> 4) << 3) * BROW];
#pragma unroll
    for (int s = 0; s < 4; ++s) {
      v16bf bmv;
      for (int e = 0; e < 8; ++e) { bmv[e] = brow[e * BROW + s * 16]; bmv[e + 8] = brow[(16 + e) * BROW + s * 16]; }
      acc[s] = wmma_bf16(a, bmv, acc[s]);
    }
  }
  int half = lane >> 4, col = lane & 15;
#pragma unroll
  for (int s = 0; s < 4; ++s)
#pragma unroll
    for (int r = 0; r < 8; ++r)
      out[(size_t)(m0 + r + 8 * half) * CC + n0 + s * 16 + col] = acc[s][r];
}

extern "C" void kernel_launch(void* const* d_in, const int* in_sizes, int n_in,
                              void* d_out, int out_size, void* d_ws, size_t ws_size,
                              hipStream_t stream) {
  const float* x      = (const float*)d_in[0];
  const float* w_attn = (const float*)d_in[1];
  const float* w_proj = (const float*)d_in[2];
  float* out = (float*)d_out;

  char* ws = (char*)d_ws;
  size_t off = 0;
  auto walloc = [&](size_t bytes) { char* p = ws + off; off += (bytes + 255) & ~(size_t)255; return p; };
  bf16* xb  = (bf16*)walloc((size_t)BB * TT * CC * 2);
  bf16* wab = (bf16*)walloc((size_t)CC * C3 * 2);
  bf16* wpb = (bf16*)walloc((size_t)CC * CC * 2);
  bf16* qb  = (bf16*)walloc((size_t)BB * HH * TT * DD * 2);
  bf16* ktb = (bf16*)walloc((size_t)BB * HH * DD * TT * 2);
  bf16* vb  = (bf16*)walloc((size_t)BB * HH * TT * DD * 2);
  bf16* yb  = (bf16*)walloc((size_t)BB * TT * CC * 2);

  f32_to_bf16<<<1024, 256, 0, stream>>>(x,      xb,  BB * TT * CC);
  f32_to_bf16<<<1024, 256, 0, stream>>>(w_attn, wab, CC * C3);
  f32_to_bf16<<<1024, 256, 0, stream>>>(w_proj, wpb, CC * CC);

  // QKV: (M/128)*(3C/64) = 64*48 blocks, 8 waves each
  qkv_gemm<<<64 * 48, 256, 0, stream>>>(xb, wab, qb, ktb, vb);
  // Attention: B*H*(T/16) = 8192 waves, 4 waves/block
  attn_kernel<<<(BB * HH * (TT / 16)) / 4, 128, 0, stream>>>(qb, ktb, vb, yb);
  // Projection: (M/128)*(C/64) = 64*16 blocks
  proj_gemm<<<64 * 16, 256, 0, stream>>>(yb, wpb, out);
}
